// SceneConstructionModel_8581344657622
// MI455X (gfx1250) — matrix operand order repaired and checked by reading, not compiled, not run on hardware
//
#include <hip/hip_runtime.h>
#include <hip/hip_bf16.h>

#define E_TOTAL 65536
#define D_FEAT  512
#define K_TOTAL 1024   // 2*D
#define H_DIM   256
#define NREL    4

typedef __attribute__((ext_vector_type(16))) __bf16 v16bf;
typedef __attribute__((ext_vector_type(8)))  float  v8f;

union BF16Frag {
    v16bf v;
    unsigned int u[8];
    uint4 q[2];
};

__device__ __forceinline__ unsigned short f2bf_rne(float f) {
    unsigned int u = __float_as_uint(f);
    u += 0x7FFFu + ((u >> 16) & 1u);       // round-to-nearest-even
    return (unsigned short)(u >> 16);
}

// Pack two f32 -> two bf16 (truncate) in ONE v_perm_b32:
// result = { hi[31:24], hi[23:16], lo[31:24], lo[23:16] } (LE bytes 3..0)
__device__ __forceinline__ unsigned int pack2bf(float lo, float hi) {
    return __builtin_amdgcn_perm(__float_as_uint(hi), __float_as_uint(lo),
                                 0x07060302u);
}

// ---------------------------------------------------------------------------
// Prep: collapse the linear decoder.  WeffT[ch][f] = sum_j W1[f][j]*W2[j][ch]
// (bf16 RNE, rows 4..15 zero), beff[ch] = b2[ch] + sum_j b1[j]*W2[j][ch].
// ---------------------------------------------------------------------------
__global__ __launch_bounds__(256) void prep_weff(
    const float* __restrict__ W1, const float* __restrict__ b1,
    const float* __restrict__ W2, const float* __restrict__ b2,
    unsigned short* __restrict__ weffT, float* __restrict__ beff)
{
    const int f = blockIdx.x * 256 + threadIdx.x;   // 0..1023
    float a0 = 0.f, a1 = 0.f, a2 = 0.f, a3 = 0.f;
    for (int j = 0; j < H_DIM; ++j) {
        const float  w  = W1[f * H_DIM + j];
        const float4 w2 = ((const float4*)W2)[j];   // W2 row j: 4 channels
        a0 += w * w2.x; a1 += w * w2.y; a2 += w * w2.z; a3 += w * w2.w;
    }
    weffT[0 * K_TOTAL + f] = f2bf_rne(a0);
    weffT[1 * K_TOTAL + f] = f2bf_rne(a1);
    weffT[2 * K_TOTAL + f] = f2bf_rne(a2);
    weffT[3 * K_TOTAL + f] = f2bf_rne(a3);
    #pragma unroll
    for (int ch = NREL; ch < 16; ++ch) weffT[ch * K_TOTAL + f] = 0;

    if (f < NREL) {
        float acc = b2[f];
        for (int j = 0; j < H_DIM; ++j) acc += b1[j] * W2[j * NREL + f];
        beff[f] = acc;
    }
}

// ---------------------------------------------------------------------------
// Main kernel: out[e][0..3] = sigmoid( concat(x[src],x[dst]) . Weff + beff )
// One wave per 16-edge tile.  D(16x16) = A(16x32 bf16, WeffT from LDS)
//                                      x B(32x16 bf16, gathered features).
// K loop split into src half / dst half (immediate-offset addressing) and
// two accumulator chains (hides the WMMA->WMMA hazard NOPs).
// ---------------------------------------------------------------------------
__global__ __launch_bounds__(256) void edge_mlp_wmma(
    const float* __restrict__ x, const int* __restrict__ ei,
    const unsigned short* __restrict__ weffT, const float* __restrict__ beff,
    float* __restrict__ out)
{
    __shared__ unsigned short ldsW[16 * K_TOTAL];   // 32 KB

    // Stage WeffT into LDS (b128 copies, all 256 threads).
    {
        const uint4* g = (const uint4*)weffT;
        uint4*       l = (uint4*)ldsW;
        #pragma unroll
        for (int i = threadIdx.x; i < (16 * K_TOTAL) / 8; i += 256) l[i] = g[i];
    }
    __syncthreads();

    const int lane = threadIdx.x & 31;
    const int wave = threadIdx.x >> 5;
    const int r    = lane & 15;        // A row (out channel) / B column (edge)
    const int g16  = lane >> 4;        // lane group 0/1
    const int tileBase = blockIdx.x * 128 + wave * 16;
    const int e    = tileBase + r;

    const int srcN = ei[e];
    const int dstN = ei[E_TOTAL + e];
    const float* srcRow = x + (size_t)srcN * D_FEAT;
    const float* dstRow = x + (size_t)dstN * D_FEAT;
    __builtin_prefetch(srcRow, 0, 0);
    __builtin_prefetch(dstRow, 0, 0);

    const unsigned short* lrow = ldsW + r * K_TOTAL + g16 * 16;
    v8f acc0 = {}, acc1 = {};

    #pragma unroll
    for (int half = 0; half < 2; ++half) {
        // Fixed base pointers -> global_load_b128 with immediate offsets.
        const float4* bp = (const float4*)((half ? dstRow : srcRow) + g16 * 16);
        const unsigned short* arow = lrow + half * D_FEAT;

        #pragma unroll 4
        for (int ks = 0; ks < D_FEAT / 32; ++ks) {
            const int k0 = ks * 32;

            // A fragment: v[0..15] <- K = k0 + g*16 + [0..16)
            BF16Frag a;
            a.q[0] = *(const uint4*)(arow + k0);
            a.q[1] = *(const uint4*)(arow + k0 + 8);

            // B fragment: lane group g holds K = k0 + g*16 + [0..16) of col e.
            const float4 f0 = bp[ks * 8 + 0];
            const float4 f1 = bp[ks * 8 + 1];
            const float4 f2 = bp[ks * 8 + 2];
            const float4 f3 = bp[ks * 8 + 3];
            BF16Frag b;
            b.u[0] = pack2bf(f0.x, f0.y); b.u[1] = pack2bf(f0.z, f0.w);
            b.u[2] = pack2bf(f1.x, f1.y); b.u[3] = pack2bf(f1.z, f1.w);
            b.u[4] = pack2bf(f2.x, f2.y); b.u[5] = pack2bf(f2.z, f2.w);
            b.u[6] = pack2bf(f3.x, f3.y); b.u[7] = pack2bf(f3.z, f3.w);

            if (ks & 1)
                acc1 = __builtin_amdgcn_wmma_f32_16x16x32_bf16(
                           false, a.v, false, b.v, (short)0, acc1, false, false);
            else
                acc0 = __builtin_amdgcn_wmma_f32_16x16x32_bf16(
                           false, a.v, false, b.v, (short)0, acc0, false, false);
        }
    }

    const v8f acc = acc0 + acc1;

    // D: VGPR i, lanes 0-15 hold row i (= channel), col = lane (= edge).
    if (lane < 16) {
        const float4 bb = *(const float4*)beff;
        float4 o;
        o.x = 1.f / (1.f + __expf(-(acc[0] + bb.x)));
        o.y = 1.f / (1.f + __expf(-(acc[1] + bb.y)));
        o.z = 1.f / (1.f + __expf(-(acc[2] + bb.z)));
        o.w = 1.f / (1.f + __expf(-(acc[3] + bb.w)));
        ((float4*)out)[tileBase + lane] = o;
    }
}

// ---------------------------------------------------------------------------
extern "C" void kernel_launch(void* const* d_in, const int* in_sizes, int n_in,
                              void* d_out, int out_size, void* d_ws, size_t ws_size,
                              hipStream_t stream) {
    const float* x  = (const float*)d_in[0];
    const int*   ei = (const int*)d_in[1];
    // d_in[2] = edge_features: unused by the identity encoder
    const float* W1 = (const float*)d_in[3];
    const float* b1 = (const float*)d_in[4];
    const float* W2 = (const float*)d_in[5];
    const float* b2 = (const float*)d_in[6];

    unsigned short* weffT = (unsigned short*)d_ws;                       // 32 KB
    float*          beff  = (float*)((char*)d_ws + 16 * K_TOTAL * 2);    // 16 B

    prep_weff<<<K_TOTAL / 256, 256, 0, stream>>>(W1, b1, W2, b2, weffT, beff);
    edge_mlp_wmma<<<E_TOTAL / 128, 256, 0, stream>>>(x, ei, weffT, beff, (float*)d_out);
}
